// StackedLSTM_18442589570255
// MI455X (gfx1250) — matrix operand order repaired and checked by reading, not compiled, not run on hardware
//
#include <hip/hip_runtime.h>
#include <hip/hip_bf16.h>

// ---------------------------------------------------------------------------
// Stacked depth-3 tanh-RNN, B=16, T=2048, H=256 on MI455X (gfx1250).
// Latency-bound sequential scan -> 3-WGP layer pipeline, register-resident
// bf16 weight fragments, dual-chain v_wmma_f32_16x16x32_bf16 inner kernel,
// bank-conflict-free padded LDS activation buffer, async global<->LDS
// inter-layer handoff (ASYNCcnt path).
// ---------------------------------------------------------------------------

#define Hdim    256
#define Bdim    16
#define Tdim    2048
#define NDEPTH  3
#define NSLOT   16          // ring slots per inter-layer link
#define THREADS 512         // 16 waves of 32
#define LDSTR   520         // activation row stride (elems): 512 + 8 pad
                            // -> 260 dwords/row = 4 (mod 64) banks: 16 rows
                            //    hit 16 disjoint 4-bank groups (no conflicts)

typedef __attribute__((ext_vector_type(16))) __bf16 bf16x16;
typedef __attribute__((ext_vector_type(8)))  __bf16 bf16x8;
typedef __attribute__((ext_vector_type(8)))  float  f32x8;

// ---------------------------------------------------------------------------
// Pack fp32 weights into WMMA B-fragment order (bf16), W_ih and W_hh fused
// into one K=512 run per output tile (ktot 0..7 = W_ih, 8..15 = W_hh).
// B-matrix (32x16, KxN) layout chosen here (contiguous k, as ldfrag reads A) per wave32 lane:
//   lane l: N = l%16 ;  element j (0..15): K = (l/16)*16 + j  within a k-tile
// Packed element index:
//   p = (((layer*16 + nt)*16 + ktot)*32 + lane)*16 + j
// Also zero-initializes the pipeline flags each call (determinism).
// ---------------------------------------------------------------------------
__global__ __launch_bounds__(256)
void rnn_pack_weights(const float* __restrict__ W_ih,
                      const float* __restrict__ W_hh,
                      __bf16* __restrict__ packed,
                      int* __restrict__ flags)
{
    if (blockIdx.x == 0 && threadIdx.x < 8) flags[threadIdx.x] = 0;

    int p     = blockIdx.x * 256 + threadIdx.x;   // 3*16*16*32*16 = 393216
    int j     = p & 15;
    int lane  = (p >> 4) & 31;
    int ktot  = (p >> 9) & 15;
    int nt    = (p >> 13) & 15;
    int layer = p >> 17;

    int mat = ktot >> 3;                      // 0 = W_ih, 1 = W_hh
    int N   = nt * 16 + (lane & 15);
    int K   = (ktot & 7) * 32 + (lane >> 4) * 16 + j;

    const float* W = mat ? W_hh : W_ih;
    float w = W[((size_t)layer * Hdim + N) * Hdim + K];   // B[k][n] = W[n][k]
    packed[p] = (__bf16)w;
}

// ---------------------------------------------------------------------------
// Main persistent pipeline kernel: gridDim.x == 3 (one block per layer).
// flags: [ready_link0, done_link0, ready_link1, done_link1, ...pad]
// links: [2][NSLOT][16*256] bf16 activation slots.
// ---------------------------------------------------------------------------
__global__ __launch_bounds__(THREADS)
void rnn_pipeline(const float* __restrict__ x,
                  const int*   __restrict__ seq_lens,
                  const float* __restrict__ bias,
                  const __bf16* __restrict__ packed,
                  float* __restrict__ out,
                  __bf16* __restrict__ links,
                  int* flags)
{
    const int layer = blockIdx.x;
    const int tid   = threadIdx.x;
    const int lane  = tid & 31;
    const int nt    = tid >> 5;             // wave id == output column tile
    const int col   = lane & 15;
    const int colg  = nt * 16 + col;
    const int ksel  = (lane >> 4) * 16;     // A-fragment K sub-block select
    const int mbase = (lane >> 4) * 8;      // C/D row base (ISA C/D layout)
    const int arow  = lane & 15;            // A-fragment M row
    const int row   = tid >> 5;             // copy-phase row
    const int cidx  = tid & 31;             // copy-phase 16B chunk in row

    // activation buffer: cols [0,256) = layer input, [256,512) = h[t-1]
    __shared__ __bf16 lds_act[Bdim * LDSTR];   // 16.25 KB
    __shared__ int    lds_slen[Bdim];

    // LDS byte offsets for the async data movers (flat LDS addr[31:0] = offset)
    const unsigned int l_in =
        (unsigned int)(uintptr_t)(lds_act + row * LDSTR) + (unsigned int)cidx * 16u;
    const unsigned int l_h =
        (unsigned int)(uintptr_t)(lds_act + row * LDSTR + 256) + (unsigned int)cidx * 16u;

    // ---- register-resident weight fragments (128 VGPRs per wave) ----
    bf16x16 w[16];
    const bf16x16* pk = reinterpret_cast<const bf16x16*>(packed);
    #pragma unroll
    for (int k2 = 0; k2 < 16; ++k2)
        w[k2] = pk[((layer * 16 + nt) * 16 + k2) * 32 + lane];
    const float bia = bias[layer * Hdim + colg];

    if (tid < Bdim) lds_slen[tid] = seq_lens[tid];
    // h[-1] = 0  (zero the h half of each row)
    reinterpret_cast<uint4*>(lds_act + row * LDSTR + 256)[cidx] =
        make_uint4(0u, 0u, 0u, 0u);
    __syncthreads();

    union AF { bf16x16 v; bf16x8 h[2]; };
    auto ldfrag = [&](int k2) -> bf16x16 {
        AF a;
        const __bf16* base = lds_act + arow * LDSTR + k2 * 32 + ksel;
        a.h[0] = *reinterpret_cast<const bf16x8*>(base);
        a.h[1] = *reinterpret_cast<const bf16x8*>(base + 8);
        return a.v;
    };

    for (int t = 0; t < Tdim; ++t) {
        // ------------- acquire input tile into lds_act[:, 0:256) -------------
        if (layer == 0) {
            const int c0 = cidx * 8;
            const float* xs = x + ((size_t)row * Tdim + t) * Hdim + c0;
            float4 a = *reinterpret_cast<const float4*>(xs);
            float4 b = *reinterpret_cast<const float4*>(xs + 4);
            __bf16* d = lds_act + row * LDSTR + c0;
            d[0] = (__bf16)a.x; d[1] = (__bf16)a.y;
            d[2] = (__bf16)a.z; d[3] = (__bf16)a.w;
            d[4] = (__bf16)b.x; d[5] = (__bf16)b.y;
            d[6] = (__bf16)b.z; d[7] = (__bf16)b.w;
            if (t + 1 < Tdim) __builtin_prefetch(xs + Hdim, 0, 1);
        } else {
            int* ready_in = &flags[(layer - 1) * 2 + 0];
            if (tid == 0) {
                while (__hip_atomic_load(ready_in, __ATOMIC_ACQUIRE,
                                         __HIP_MEMORY_SCOPE_AGENT) < t + 1)
                    __builtin_amdgcn_s_sleep(1);
            }
            __syncthreads();
            const __bf16* slot = links
                + ((size_t)(layer - 1) * NSLOT + (t & (NSLOT - 1))) * (Bdim * Hdim);
            // async L2 -> LDS, device scope (bypasses possibly-stale WGP$)
            unsigned long long g =
                (unsigned long long)(uintptr_t)slot + (unsigned long long)tid * 16ull;
            asm volatile("global_load_async_to_lds_b128 %0, %1, off scope:SCOPE_DEV"
                         :: "v"(l_in), "v"(g) : "memory");
            asm volatile("s_wait_asynccnt 0x0" ::: "memory");
        }
        __syncthreads();
        if (layer > 0 && tid == 0) {
            int* done_in = &flags[(layer - 1) * 2 + 1];
            __hip_atomic_store(done_in, t + 1, __ATOMIC_RELEASE,
                               __HIP_MEMORY_SCOPE_AGENT);
        }

        // ------------- dual-chain WMMA over fused K=512 -------------
        f32x8 acc0 = {}, acc1 = {};
        bf16x16 cur0 = ldfrag(0);
        bf16x16 cur1 = ldfrag(1);
        #pragma unroll
        for (int k2 = 0; k2 < 16; k2 += 2) {
            bf16x16 n0 = (k2 + 2 < 16) ? ldfrag(k2 + 2) : cur0;
            bf16x16 n1 = (k2 + 3 < 16) ? ldfrag(k2 + 3) : cur1;
            acc0 = __builtin_amdgcn_wmma_f32_16x16x32_bf16(
                       false, cur0, false, w[k2],     (short)0, acc0, false, false);
            acc1 = __builtin_amdgcn_wmma_f32_16x16x32_bf16(
                       false, cur1, false, w[k2 + 1], (short)0, acc1, false, false);
            cur0 = n0; cur1 = n1;
        }

        // fast tanh: tanh(s) = 1 - 2/(1 + 2^(2*log2e*s)); saturates at +-1
        float hv[8];
        #pragma unroll
        for (int i = 0; i < 8; ++i) {
            float s = acc0[i] + acc1[i] + bia;
            float e = __builtin_amdgcn_exp2f(s * 2.885390081777927f);
            hv[i] = 1.0f - 2.0f * __builtin_amdgcn_rcpf(1.0f + e);
        }

        __syncthreads();   // all fragment reads of lds_act done

        // ------------- h state writeback (+ producer backpressure) -------------
        #pragma unroll
        for (int i = 0; i < 8; ++i) {
            const int m = mbase + i;                       // batch row (C/D layout)
            lds_act[m * LDSTR + 256 + colg] = (__bf16)hv[i];
        }
        if (layer < 2 && tid == 0 && t >= NSLOT) {
            int* done_out = &flags[layer * 2 + 1];
            while (__hip_atomic_load(done_out, __ATOMIC_ACQUIRE,
                                     __HIP_MEMORY_SCOPE_AGENT) < t - NSLOT + 1)
                __builtin_amdgcn_s_sleep(1);
        }

        // ------------- publish to next layer (before output drain) -------------
        if (layer < 2) {
            __syncthreads();   // h region fully written
            __bf16* slot = links
                + ((size_t)layer * NSLOT + (t & (NSLOT - 1))) * (Bdim * Hdim);
            // async LDS -> L2, device scope; waited => no threadfence needed
            unsigned long long g =
                (unsigned long long)(uintptr_t)slot + (unsigned long long)tid * 16ull;
            asm volatile("global_store_async_from_lds_b128 %0, %1, off scope:SCOPE_DEV"
                         :: "v"(g), "v"(l_h) : "memory");
            asm volatile("s_wait_asynccnt 0x0" ::: "memory");
            __syncthreads();
            if (tid == 0) {
                int* ready_out = &flags[layer * 2 + 0];
                __hip_atomic_store(ready_out, t + 1, __ATOMIC_RELEASE,
                                   __HIP_MEMORY_SCOPE_AGENT);
            }
        }

        // ------------- masked output stores (off critical path) -------------
        #pragma unroll
        for (int i = 0; i < 8; ++i) {
            const int m = mbase + i;
            const float o = (t < lds_slen[m]) ? hv[i] : 0.0f;
            out[(((size_t)m * Tdim + t) * NDEPTH + layer) * Hdim + colg] = o;
        }
        // for layer 2 the next iteration's sync1 orders h-writes vs. reads;
        // for producers the post-publish barrier already did.
        if (layer == 2) __syncthreads();
    }
}

// ---------------------------------------------------------------------------
// Workspace layout (needs ~1.06 MB):
//   [0,            786432)  packed bf16 weight fragments (3*2*256*256)
//   [786432,      1048576)  inter-layer ring buffers 2*NSLOT*16*256 bf16
//   [1048576,     1048608)  flags (8 ints)
// ---------------------------------------------------------------------------
extern "C" void kernel_launch(void* const* d_in, const int* in_sizes, int n_in,
                              void* d_out, int out_size, void* d_ws, size_t ws_size,
                              hipStream_t stream) {
    const float* x        = (const float*)d_in[0];
    const int*   seq_lens = (const int*)  d_in[1];
    const float* W_ih     = (const float*)d_in[2];
    const float* W_hh     = (const float*)d_in[3];
    const float* bias     = (const float*)d_in[4];
    float* out = (float*)d_out;

    char* ws = (char*)d_ws;
    __bf16* packed = (__bf16*)(ws);
    __bf16* links  = (__bf16*)(ws + 786432);
    int*    flags  = (int*)   (ws + 786432 + 2 * NSLOT * Bdim * Hdim * 2);

    // Repack weights + reset pipeline flags (runs every call: deterministic).
    rnn_pack_weights<<<1536, 256, 0, stream>>>(W_ih, W_hh, packed, flags);

    // One workgroup per layer, pipelined over t.
    rnn_pipeline<<<NDEPTH, THREADS, 0, stream>>>(x, seq_lens, bias, packed,
                                                 out, links, flags);
}
